// WiseDiffModule_78615081386009
// MI455X (gfx1250) — matrix operand order repaired and checked by reading, not compiled, not run on hardware
//
#include <hip/hip_runtime.h>
#include <hip/hip_bf16.h>

typedef __attribute__((ext_vector_type(16))) _Float16 v16h;
typedef __attribute__((ext_vector_type(8)))  _Float16 v8h;
typedef __attribute__((ext_vector_type(8)))  float    v8f;

#define BDIM   256
#define BB     256
#define CC     512
#define HW     49
#define NPAD   64
#define KDIM   1024
#define ODIM   512
#define LDS_PITCH 2064   // 64 n-rows * (2048B data + 16B pad) -> conflict-free ds_load_b128

// ---------------- Phase 0: conv_w f32 -> f16 ----------------
__global__ void cvt_w_kernel(const float* __restrict__ cw, _Float16* __restrict__ cw16, int n) {
    int i = blockIdx.x * blockDim.x + threadIdx.x;
    if (i < n) cw16[i] = (_Float16)cw[i];
}

// ---------------- Phase 1: bilateral filter + pack F1^T (f16) ----------------
// F1t layout: [b][n=64 (hw, zero-padded 49..63)][k=1024 (c: V 0..511, Y 512..1023)]
__global__ void bilateral_pack_kernel(const float* __restrict__ T, const float* __restrict__ V,
                                      const float* __restrict__ W9, _Float16* __restrict__ F1t) {
    extern __shared__ float sm[];          // 2 * 256*49 floats = 100352 bytes
    float* sT = sm;
    float* sV = sm + 256 * HW;
    const int tid = threadIdx.x;
    const int b   = blockIdx.x >> 1;
    const int cg  = blockIdx.x & 1;
    const size_t base = ((size_t)b * CC + (size_t)cg * 256) * HW;

    for (int i = tid; i < 256 * HW; i += BDIM) {   // coalesced LDS fill
        sT[i] = T[base + i];
        sV[i] = V[base + i];
    }
    float w[9];
#pragma unroll
    for (int k = 0; k < 9; ++k) w[k] = W9[k];
    __syncthreads();

    const float* tI = &sT[tid * HW];   // per-thread image; stride 49 is conflict-free mod 64 banks
    const float* vI = &sV[tid * HW];
    const int kcol = cg * 256 + tid;                     // channel index in [0,512)
    _Float16* outV = F1t + (size_t)b * NPAD * KDIM + kcol;
    _Float16* outY = outV + 512;

    for (int cell = 0; cell < HW; ++cell) {
        const int h = cell / 7, wq = cell % 7;
        const float tc = tI[cell], vc = vI[cell];
        float acc = 0.f;
#pragma unroll
        for (int k = 0; k < 9; ++k) {
            int hh = h + (k / 3) - 1; hh = (hh < 0) ? 1 : (hh > 6 ? 5 : hh);  // reflect pad
            int ww = wq + (k % 3) - 1; ww = (ww < 0) ? 1 : (ww > 6 ? 5 : ww);
            const float dT = tI[hh * 7 + ww] - tc;
            const float dV = vI[hh * 7 + ww] - vc;
            acc += w[k] * __expf(-dT * dT * 0.25f) * dV;   // SIGMA=2 -> /4
        }
        outV[(size_t)cell * KDIM] = (_Float16)vc;          // coalesced across threads
        outY[(size_t)cell * KDIM] = (_Float16)acc;
    }
    for (int cell = HW; cell < NPAD; ++cell) {             // zero-pad N to 64
        outV[(size_t)cell * KDIM] = (_Float16)0.f;
        outY[(size_t)cell * KDIM] = (_Float16)0.f;
    }
}

// ---------------- Phase 2: batched GEMM via WMMA f16 -> f32 ----------------
// grid: 1024 blocks = 256 batches x 4 M-groups; 256 threads = 8 waves; wave = 16(M) x 64(N)
__global__ void gemm_kernel(const _Float16* __restrict__ F1t, const _Float16* __restrict__ cw16,
                            const float* __restrict__ convb, float* __restrict__ out) {
    extern __shared__ char smem[];   // 64 * LDS_PITCH = 132096 bytes (B slab, padded rows)
    const int tid      = threadIdx.x;
    const int b        = blockIdx.x >> 2;
    const int mg       = blockIdx.x & 3;
    const int wv       = tid >> 5;
    const int lane     = tid & 31;
    const int laneHalf = lane >> 4;        // 0 | 1
    const int lane16   = lane & 15;
    const int m_base   = mg * 128 + wv * 16;

    // Async copy F1t[b] (64 rows x 2048B) into LDS with +16B/row pad (ASYNCcnt path)
    {
        const uint32_t ldsbase = (uint32_t)(uintptr_t)smem;
        const uint64_t gbase   = (uint64_t)(uintptr_t)(F1t + (size_t)b * NPAD * KDIM);
        for (int idx = tid; idx < 64 * 128; idx += BDIM) {
            const int row = idx >> 7, chunk = idx & 127;
            const uint32_t la = ldsbase + (uint32_t)(row * LDS_PITCH + chunk * 16);
            const uint64_t ga = gbase + (uint64_t)(row * 2048 + chunk * 16);
            asm volatile("global_load_async_to_lds_b128 %0, %1, off"
                         :: "v"(la), "v"(ga) : "memory");
        }
        asm volatile("s_wait_asynccnt 0" ::: "memory");
    }
    __syncthreads();

    v8f acc[4] = {v8f{}, v8f{}, v8f{}, v8f{}};
    const _Float16* aRow = cw16 + (size_t)(m_base + lane16) * KDIM;

    for (int kt = 0; kt < 32; ++kt) {
        const int kb = kt * 32;
        // A fragment (16x32 f16): lane<16 holds K {kb..kb+7, kb+16..23}; lane>=16 the +8 halves
        const _Float16* ap = aRow + kb + laneHalf * 8;
        const v8h alo = *(const v8h*)(ap);
        const v8h ahi = *(const v8h*)(ap + 16);
        v16h a;
#pragma unroll
        for (int i = 0; i < 8; ++i) { a[i] = alo[i]; a[8 + i] = ahi[i]; }

#pragma unroll
        for (int nt = 0; nt < 4; ++nt) {
            // B fragment (32x16 f16): N = lane16, K {kk..kk+7, kk+16..kk+23} matching A
            const int n  = nt * 16 + lane16;
            const int kk = kb + laneHalf * 8;
            const char* bp = smem + n * LDS_PITCH + kk * 2;
            const v8h blo = *(const v8h*)(bp);
            const v8h bhi = *(const v8h*)(bp + 32);
            v16h bm;
#pragma unroll
            for (int i = 0; i < 8; ++i) { bm[i] = blo[i]; bm[8 + i] = bhi[i]; }
            acc[nt] = __builtin_amdgcn_wmma_f32_16x16x32_f16(
                false, a, false, bm, (short)0, acc[nt], false, false);
        }
    }

    // Epilogue: C/D layout -> out[b][o][hw], mask padded cols, add bias
#pragma unroll
    for (int nt = 0; nt < 4; ++nt) {
        const int col = nt * 16 + lane16;
        if (col < HW) {
#pragma unroll
            for (int r = 0; r < 8; ++r) {
                const int row_m = m_base + r + laneHalf * 8;
                out[((size_t)b * ODIM + row_m) * HW + col] = acc[nt][r] + convb[row_m];
            }
        }
    }
}

// ---------------- launch ----------------
extern "C" void kernel_launch(void* const* d_in, const int* in_sizes, int n_in,
                              void* d_out, int out_size, void* d_ws, size_t ws_size,
                              hipStream_t stream) {
    const float* T   = (const float*)d_in[0];
    const float* V   = (const float*)d_in[1];
    const float* W9  = (const float*)d_in[2];
    const float* cw  = (const float*)d_in[3];
    const float* cb  = (const float*)d_in[4];
    float* out = (float*)d_out;

    _Float16* F1t  = (_Float16*)d_ws;                          // 256*64*1024*2 = 33554432 B
    _Float16* cw16 = (_Float16*)((char*)d_ws + (size_t)33554432);  // 512*1024*2 = 1048576 B

    (void)in_sizes; (void)n_in; (void)out_size; (void)ws_size;

    hipFuncSetAttribute((const void*)bilateral_pack_kernel,
                        hipFuncAttributeMaxDynamicSharedMemorySize, 100352);
    hipFuncSetAttribute((const void*)gemm_kernel,
                        hipFuncAttributeMaxDynamicSharedMemorySize, 64 * LDS_PITCH);

    cvt_w_kernel<<<(ODIM * KDIM + BDIM - 1) / BDIM, BDIM, 0, stream>>>(cw, cw16, ODIM * KDIM);
    bilateral_pack_kernel<<<BB * 2, BDIM, 100352, stream>>>(T, V, W9, F1t);
    gemm_kernel<<<BB * 4, BDIM, 64 * LDS_PITCH, stream>>>(F1t, cw16, cb, out);
}
